// H2GNNClassifier_81518479278625
// MI455X (gfx1250) — compile-verified, matrix operand order checked by reading, not run on hardware
//
#include <hip/hip_runtime.h>
#include <hip/hip_bf16.h>

#define NNODES 100000
#define NEDGES 1600000
#define INC    128
#define HID    96
#define NGRAPH 512

typedef __attribute__((ext_vector_type(16))) __bf16 v16bf;
typedef __attribute__((ext_vector_type(8)))  float  v8f;

// ---------------- utility fills ----------------
__global__ void fill_f32(float* __restrict__ p, float v, int n) {
    int i = blockIdx.x * blockDim.x + threadIdx.x;
    if (i < n) p[i] = v;
}

// ---------------- degree / norm ----------------
__global__ void deg_accum(const int* __restrict__ dst, float* __restrict__ deg, int e) {
    int i = blockIdx.x * blockDim.x + threadIdx.x;
    if (i < e) atomicAdd(&deg[dst[i]], 1.0f);
}

__global__ void deg_to_dis(float* __restrict__ d, int n) {
    int i = blockIdx.x * blockDim.x + threadIdx.x;
    if (i < n) {
        float deg = d[i];
        d[i] = (deg > 0.0f) ? rsqrtf(fmaxf(deg, 1.0f)) : 0.0f;
    }
}

__global__ void edge_norm(const int* __restrict__ src, const int* __restrict__ dst,
                          const float* __restrict__ dis, float* __restrict__ nrm, int e) {
    int i = blockIdx.x * blockDim.x + threadIdx.x;
    if (i < e) nrm[i] = dis[src[i]] * dis[dst[i]];
}

// ---------------- bf16 WMMA GEMM: A[MxK] @ B[KxNcols] -> C row-major f32 ----------------
// grid = (ceil(tilesM/8), tilesN); block = 256 (8 waves, one 16x16 tile each, same tn).
// B strip (K x 16) is converted to bf16 and pre-swizzled into the WMMA operand layout in
// LDS once per block; each lane then reads its v16bf fragment as 32 contiguous LDS bytes.
// A is read with float4 (global_load_b128) since each lane's 16 elements are two
// contiguous 8-float runs (kbase..kbase+7, kbase+16..kbase+23).
template <int K>
__global__ void gemm_wmma_bf16(const float* __restrict__ A,
                               const float* __restrict__ B, int ldb,
                               float* __restrict__ C, int ldc,
                               int M) {
    constexpr int STEPS = K / 32;
    __shared__ __bf16 lds_b[STEPS][32][16];

    const int tn = blockIdx.y;

    // cooperative pack of the B strip into WMMA B-operand layout
    for (int t = threadIdx.x; t < STEPS * 32 * 16; t += blockDim.x) {
        const int e    = t & 15;
        const int lane = (t >> 4) & 31;
        const int step = t >> 9;
        const int p    = e >> 1;
        const int kk   = step * 32 + ((p & 3) << 1) + ((p >> 2) << 4)
                       + ((lane >> 4) << 3) + (e & 1);
        const int col  = tn * 16 + (lane & 15);
        lds_b[step][lane][e] = (__bf16)B[(size_t)kk * ldb + col];
    }
    __syncthreads();

    const int tm = blockIdx.x * (blockDim.x >> 5) + (threadIdx.x >> 5);
    if (tm >= (M >> 4)) return;                       // wave-uniform

    const int lane = threadIdx.x & 31;
    const int half = lane >> 4;
    const int idx  = lane & 15;
    const int arow = tm * 16 + idx;
    const float* __restrict__ arowp = A + (size_t)arow * K;

    v16bf bfrag[STEPS];
#pragma unroll
    for (int s = 0; s < STEPS; ++s)
        bfrag[s] = *(const v16bf*)&lds_b[s][lane][0];  // 32B contiguous -> ds_load_b128 x2

    v8f c = {};
#pragma unroll
    for (int s = 0; s < STEPS; ++s) {
        const int kbase = s * 32 + half * 8;
        const float4 a0 = *(const float4*)(arowp + kbase);
        const float4 a1 = *(const float4*)(arowp + kbase + 4);
        const float4 a2 = *(const float4*)(arowp + kbase + 16);
        const float4 a3 = *(const float4*)(arowp + kbase + 20);
        v16bf a;
        a[0]  = (__bf16)a0.x; a[1]  = (__bf16)a0.y; a[2]  = (__bf16)a0.z; a[3]  = (__bf16)a0.w;
        a[4]  = (__bf16)a1.x; a[5]  = (__bf16)a1.y; a[6]  = (__bf16)a1.z; a[7]  = (__bf16)a1.w;
        a[8]  = (__bf16)a2.x; a[9]  = (__bf16)a2.y; a[10] = (__bf16)a2.z; a[11] = (__bf16)a2.w;
        a[12] = (__bf16)a3.x; a[13] = (__bf16)a3.y; a[14] = (__bf16)a3.z; a[15] = (__bf16)a3.w;
        c = __builtin_amdgcn_wmma_f32_16x16x32_bf16(false, a, false, bfrag[s],
                                                    (short)0, c, false, false);
    }

#pragma unroll
    for (int r = 0; r < 8; ++r) {
        const int row = tm * 16 + r + half * 8;       // f32 C/D layout: VGPR r -> M=r+8*half
        C[(size_t)row * ldc + tn * 16 + idx] = c[r];
    }
}

// ---------------- edge message scatter: acc[dst] += hw[src] * norm ----------------
// block = 96 threads (one edge per block, one channel per thread) -> coalesced rows
__global__ void scatter_edges(const int* __restrict__ src, const int* __restrict__ dst,
                              const float* __restrict__ nrm,
                              const float* __restrict__ hw, float* __restrict__ acc) {
    const int e = blockIdx.x;
    const int c = threadIdx.x;
    const int s = src[e];
    const int d = dst[e];
    atomicAdd(&acc[(size_t)d * HID + c], hw[(size_t)s * HID + c] * nrm[e]);
}

// ---------------- fused self-loop + bias + ReLU (in place on acc) ----------------
__global__ void finalize_relu(float* __restrict__ acc, const float* __restrict__ hw,
                              const float* __restrict__ dis, const float* __restrict__ bias,
                              int n) {
    int i = blockIdx.x * blockDim.x + threadIdx.x;
    if (i >= n * HID) return;
    const int node = i / HID;
    const int c = i % HID;
    const float d = dis[node];
    const float v = acc[i] + hw[i] * d * d + bias[c];
    acc[i] = v > 0.0f ? v : 0.0f;
}

// ---------------- node logits: one wave32 per node, coalesced row read + shfl reduce ----
__global__ void node_logits_k(const float* __restrict__ h1, const float* __restrict__ h2,
                              const float* __restrict__ Wn, const float* __restrict__ bn,
                              float* __restrict__ out, int n) {
    const int wave = blockIdx.x * (blockDim.x >> 5) + (threadIdx.x >> 5);
    if (wave >= n) return;
    const int lane = threadIdx.x & 31;
    float s = 0.0f;
#pragma unroll
    for (int c0 = 0; c0 < HID; c0 += 32) {
        const int c = c0 + lane;
        s += h1[(size_t)wave * HID + c] * Wn[c];
        s += h2[(size_t)wave * HID + c] * Wn[HID + c];
    }
#pragma unroll
    for (int off = 16; off > 0; off >>= 1)
        s += __shfl_xor(s, off, 32);
    if (lane == 0) out[wave] = s + bn[0];
}

// ---------------- segment mean/max pooling ----------------
// h values are post-ReLU (>= 0) so int-punned atomicMax on f32 bits is order-correct,
// and 0-init matches reference's where(cnt>0, maxp, 0).
__global__ void pool_k(const float* __restrict__ h1, const float* __restrict__ h2,
                       const int* __restrict__ batch,
                       float* __restrict__ gsum, float* __restrict__ gmax,
                       float* __restrict__ cnt, int n) {
    int i = blockIdx.x * blockDim.x + threadIdx.x;
    if (i >= n * 2 * HID) return;
    const int node = i / (2 * HID);
    const int c = i % (2 * HID);
    const int b = batch[node];
    const float v = (c < HID) ? h1[(size_t)node * HID + c]
                              : h2[(size_t)node * HID + (c - HID)];
    atomicAdd(&gsum[(size_t)b * 2 * HID + c], v);
    atomicMax((int*)&gmax[(size_t)b * 2 * HID + c], __float_as_int(v));
    if (c == 0) atomicAdd(&cnt[b], 1.0f);
}

// ---------------- graph logits ----------------
__global__ void graph_logits_k(const float* __restrict__ gsum, const float* __restrict__ gmax,
                               const float* __restrict__ cnt,
                               const float* __restrict__ Wg, const float* __restrict__ bg,
                               float* __restrict__ out, int g) {
    int i = blockIdx.x * blockDim.x + threadIdx.x;
    if (i >= g) return;
    const float inv = 1.0f / fmaxf(cnt[i], 1.0f);
    float s = bg[0];
    for (int c = 0; c < 2 * HID; ++c) s += gsum[(size_t)i * 2 * HID + c] * inv * Wg[c];
    for (int c = 0; c < 2 * HID; ++c) s += gmax[(size_t)i * 2 * HID + c] * Wg[2 * HID + c];
    out[i] = s;
}

static inline int cdiv(long long a, long long b) { return (int)((a + b - 1) / b); }

extern "C" void kernel_launch(void* const* d_in, const int* in_sizes, int n_in,
                              void* d_out, int out_size, void* d_ws, size_t ws_size,
                              hipStream_t stream) {
    (void)in_sizes; (void)n_in; (void)out_size; (void)ws_size;
    const float* x     = (const float*)d_in[0];
    const int*   ei    = (const int*)d_in[1];
    const int*   batch = (const int*)d_in[2];
    const float* W1    = (const float*)d_in[3];
    const float* b1    = (const float*)d_in[4];
    const float* W2    = (const float*)d_in[5];
    const float* b2    = (const float*)d_in[6];
    const float* Wn    = (const float*)d_in[7];
    const float* bn    = (const float*)d_in[8];
    const float* Wg    = (const float*)d_in[9];
    const float* bg    = (const float*)d_in[10];
    float* out = (float*)d_out;

    const int* src = ei;            // edge_index row 0
    const int* dst = ei + NEDGES;   // edge_index row 1

    // workspace layout (floats)
    float* ws   = (float*)d_ws;
    float* dis  = ws;                                   // N      (deg -> dis, in place)
    float* nrm  = dis  + NNODES;                        // E
    float* hw   = nrm  + NEDGES;                        // N*H    (hw1, then hw2)
    float* bufB = hw   + (size_t)NNODES * HID;          // N*H    (acc1 -> h1)
    float* bufC = bufB + (size_t)NNODES * HID;          // N*H    (acc2 -> h2)
    float* gsum = bufC + (size_t)NNODES * HID;          // G*2H
    float* gmax = gsum + (size_t)NGRAPH * 2 * HID;      // G*2H
    float* cnt  = gmax + (size_t)NGRAPH * 2 * HID;      // G

    const int NH = NNODES * HID;
    const int tilesM = NNODES / 16;                     // 6250
    const int tilesN = HID / 16;                        // 6
    const dim3 gemm_grid(cdiv(tilesM, 8), tilesN);

    // --- GCN normalization: deg (self-loop = 1.0 init) -> dis -> per-edge norm ---
    fill_f32<<<cdiv(NNODES, 256), 256, 0, stream>>>(dis, 1.0f, NNODES);
    deg_accum<<<cdiv(NEDGES, 256), 256, 0, stream>>>(dst, dis, NEDGES);
    deg_to_dis<<<cdiv(NNODES, 256), 256, 0, stream>>>(dis, NNODES);
    edge_norm<<<cdiv(NEDGES, 256), 256, 0, stream>>>(src, dst, dis, nrm, NEDGES);

    // --- layer 1: hw1 = x @ W1 ; scatter ; bias+self-loop+ReLU -> h1 (bufB) ---
    gemm_wmma_bf16<INC><<<gemm_grid, 256, 0, stream>>>(x, W1, HID, hw, HID, NNODES);
    fill_f32<<<cdiv(NH, 256), 256, 0, stream>>>(bufB, 0.0f, NH);
    scatter_edges<<<NEDGES, HID, 0, stream>>>(src, dst, nrm, hw, bufB);
    finalize_relu<<<cdiv(NH, 256), 256, 0, stream>>>(bufB, hw, dis, b1, NNODES);

    // --- layer 2: hw2 = h1 @ W2 ; scatter ; bias+self-loop+ReLU -> h2 (bufC) ---
    gemm_wmma_bf16<HID><<<gemm_grid, 256, 0, stream>>>(bufB, W2, HID, hw, HID, NNODES);
    fill_f32<<<cdiv(NH, 256), 256, 0, stream>>>(bufC, 0.0f, NH);
    scatter_edges<<<NEDGES, HID, 0, stream>>>(src, dst, nrm, hw, bufC);
    finalize_relu<<<cdiv(NH, 256), 256, 0, stream>>>(bufC, hw, dis, b2, NNODES);

    // --- node logits (out[G:]) : wave per node ---
    node_logits_k<<<cdiv(NNODES, 8), 256, 0, stream>>>(bufB, bufC, Wn, bn,
                                                       out + NGRAPH, NNODES);

    // --- graph pooling + graph logits (out[0:G]) ---
    fill_f32<<<cdiv(NGRAPH * 2 * HID, 256), 256, 0, stream>>>(gsum, 0.0f, NGRAPH * 2 * HID);
    fill_f32<<<cdiv(NGRAPH * 2 * HID, 256), 256, 0, stream>>>(gmax, 0.0f, NGRAPH * 2 * HID);
    fill_f32<<<cdiv(NGRAPH, 256), 256, 0, stream>>>(cnt, 0.0f, NGRAPH);
    pool_k<<<cdiv((long long)NNODES * 2 * HID, 256), 256, 0, stream>>>(bufB, bufC, batch,
                                                                      gsum, gmax, cnt, NNODES);
    graph_logits_k<<<cdiv(NGRAPH, 256), 256, 0, stream>>>(gsum, gmax, cnt, Wg, bg,
                                                          out, NGRAPH);
}
